// InfiniAttentionLM_9019431322037
// MI455X (gfx1250) — matrix operand order repaired and checked by reading, not compiled, not run on hardware
//
#include <hip/hip_runtime.h>

#define S_   1024
#define B_   4
#define D_   768
#define NH_  12
#define DH_  64
#define L_   4
#define DI_  3072
#define MROWS (S_*B_)    /* 4096 */
#define KVW   (2*D_)     /* 1536 */

#define AS1 __attribute__((address_space(1)))
#define AS3 __attribute__((address_space(3)))

#if defined(__has_builtin)
#if __has_builtin(__builtin_amdgcn_global_load_async_to_lds_b128) && \
    __has_builtin(__builtin_amdgcn_s_wait_asynccnt)
#define USE_ASYNC_LDS 1
#endif
#endif

typedef __bf16 bf16_t;
typedef __attribute__((ext_vector_type(8)))  bf16_t v8bf;
typedef __attribute__((ext_vector_type(16))) bf16_t v16bf;
typedef __attribute__((ext_vector_type(8)))  float  v8f;
typedef int v4i_ld __attribute__((vector_size(16)));   // int4 for async-LDS builtin

__device__ __forceinline__ unsigned short f2bf(float f) {
  unsigned int u = __builtin_bit_cast(unsigned int, f);
  u += 0x7FFFu + ((u >> 16) & 1u);            // round-to-nearest-even
  return (unsigned short)(u >> 16);
}

__device__ __forceinline__ v16bf cat8(v8bf lo, v8bf hi) {
  return __builtin_shufflevector(lo, hi, 0,1,2,3,4,5,6,7,8,9,10,11,12,13,14,15);
}

__device__ __forceinline__ float elu1(float x) {
  return x > 0.0f ? x + 1.0f : __expf(x);
}

// ---------------------------------------------------------------- cast f32->bf16
__global__ void cast_bf16_kernel(const float* __restrict__ src,
                                 unsigned short* __restrict__ dst, int n) {
  int i = blockIdx.x * 256 + threadIdx.x;
  if (i < n) dst[i] = f2bf(src[i]);
}

// ---------------------------------------------------------------- embedding + sinusoid pos
__global__ void embed_kernel(const int* __restrict__ inp, const float* __restrict__ we,
                             float* __restrict__ hf, unsigned short* __restrict__ hb) {
  int i = blockIdx.x * 256 + threadIdx.x;          // over S*B*D
  if (i >= MROWS * D_) return;
  int d  = i % D_;
  int rb = i / D_;                                  // s*B + b
  int s  = rb / B_;
  int tok = inp[rb];
  float pos = (float)(S_ - 1 - s);
  int j = (d < D_/2) ? d : d - D_/2;
  float invf = __expf(-((float)(2*j) / (float)D_) * 9.210340371976184f); // ln(1e4)
  float si = pos * invf;
  float pe = (d < D_/2) ? sinf(si) : cosf(si);
  float v = we[(size_t)tok * D_ + d] * 27.712812921102035f + pe;   // sqrt(768)
  hf[i] = v;
  hb[i] = f2bf(v);
}

// ---------------------------------------------------------------- WMMA GEMM: C[M,N] = A[M,K] * W[N,K]^T
// A, W bf16 row-major (K contiguous). One wave computes a 32x64 tile.
__global__ void __launch_bounds__(32)
gemm_bf16_wmma(const unsigned short* __restrict__ A, const unsigned short* __restrict__ W,
               const float* __restrict__ bias, float* __restrict__ Cf,
               unsigned short* __restrict__ Cb, int M, int N, int K, int relu) {
  const int lane = threadIdx.x;
  const int l16  = lane & 15;
  const int hi   = lane >> 4;       // 0 or 1
  const int hi8  = hi * 8;
  const int m0   = blockIdx.y * 32;
  const int n0   = blockIdx.x * 64;

  v8f acc[2][4];
#pragma unroll
  for (int t = 0; t < 2; ++t)
#pragma unroll
    for (int u = 0; u < 4; ++u)
#pragma unroll
      for (int r = 0; r < 8; ++r) acc[t][u][r] = 0.0f;

  const unsigned short* aRow[2];
#pragma unroll
  for (int t = 0; t < 2; ++t) aRow[t] = A + (size_t)(m0 + t*16 + l16) * K + hi*16;
  const unsigned short* bRow[4];
#pragma unroll
  for (int u = 0; u < 4; ++u) bRow[u] = W + (size_t)(n0 + u*16 + l16) * K + hi*16;

  for (int kb = 0; kb < K; kb += 32) {
    if (kb + 32 < K) {
      __builtin_prefetch(aRow[0] + kb + 32, 0, 1);
      __builtin_prefetch(bRow[0] + kb + 32, 0, 1);
    }
    v16bf afr[2], bfr[4];
#pragma unroll
    for (int t = 0; t < 2; ++t) {
      v8bf lo = *(const v8bf*)(aRow[t] + kb);        // K = kb+hi*16 .. +7
      v8bf hh = *(const v8bf*)(aRow[t] + kb + 8);    // K = kb+hi*16+8 .. +15
      afr[t] = cat8(lo, hh);
    }
#pragma unroll
    for (int u = 0; u < 4; ++u)
      bfr[u] = *(const v16bf*)(bRow[u] + kb);        // K = kb+hi*16 .. +15 (contig)
#pragma unroll
    for (int t = 0; t < 2; ++t)
#pragma unroll
      for (int u = 0; u < 4; ++u)
        acc[t][u] = __builtin_amdgcn_wmma_f32_16x16x32_bf16(
            false, afr[t], false, bfr[u], (short)0, acc[t][u], false, false);
  }

#pragma unroll
  for (int t = 0; t < 2; ++t)
#pragma unroll
    for (int u = 0; u < 4; ++u) {
      const int col = n0 + u*16 + l16;
      float bv = bias ? bias[col] : 0.0f;
#pragma unroll
      for (int r = 0; r < 8; ++r) {
        int row = m0 + t*16 + hi8 + r;                // C layout: lanes>=16 hold M=r+8
        float v = acc[t][u][r] + bv;
        if (relu) v = fmaxf(v, 0.0f);
        size_t idx = (size_t)row * N + col;
        if (Cf) Cf[idx] = v;
        if (Cb) Cb[idx] = f2bf(v);
      }
    }
}

// ---------------------------------------------------------------- flash attention (causal)
// one wave per 16-row query block per (b, head). q rows stride B*D, kv rows stride B*KVW.
__global__ void __launch_bounds__(32)
flash_attn_wmma(const unsigned short* __restrict__ qb,
                const unsigned short* __restrict__ kvb,
                float* __restrict__ attn) {
  __shared__ __align__(32) unsigned short ldsP[16 * 32];   // P tile, row-major 16x32
#if USE_ASYNC_LDS
  __shared__ __align__(32) unsigned short ldsV[32 * 64];   // V tile row-major [j][d]
#else
  __shared__ __align__(32) unsigned short ldsVt[64 * 32];  // V^T tile: [d][j]
#endif
  const int lane = threadIdx.x;
  const int l16  = lane & 15;
  const int hi   = lane >> 4;
  const int hi8  = hi * 8;
  const int i0 = blockIdx.x * 16;
  const int b  = blockIdx.y;
  const int n  = blockIdx.z;
  const float scale = 0.125f;   // 1/sqrt(DH)

  // Q fragments (16 x 64, two K=32 chunks)
  v16bf qA[2];
  {
    const unsigned short* qp = qb + ((size_t)(i0 + l16) * B_ + b) * D_ + n*DH_ + hi*16;
#pragma unroll
    for (int c = 0; c < 2; ++c) {
      v8bf lo = *(const v8bf*)(qp + c*32);
      v8bf hh = *(const v8bf*)(qp + c*32 + 8);
      qA[c] = cat8(lo, hh);
    }
  }

  v8f o[4];
#pragma unroll
  for (int u = 0; u < 4; ++u)
#pragma unroll
    for (int r = 0; r < 8; ++r) o[u][r] = 0.0f;
  float mrow[8], lrow[8];
#pragma unroll
  for (int r = 0; r < 8; ++r) { mrow[r] = -1e30f; lrow[r] = 0.0f; }

  for (int jb = 0; jb < i0 + 16; jb += 32) {
    __syncthreads();
    // --- stage V tile (32 j-rows x 64 d) into LDS
#if USE_ASYNC_LDS
#pragma unroll
    for (int it = 0; it < 8; ++it) {
      int linear = it*32 + lane;       // 0..255
      int j  = linear >> 3;            // 0..31
      int c8 = linear & 7;             // 8-element chunk of d
      const unsigned short* vp =
          kvb + ((size_t)(jb + j) * B_ + b) * KVW + D_ + n*DH_ + c8*8;
      __builtin_amdgcn_global_load_async_to_lds_b128(
          (AS1 v4i_ld*)(void*)vp, (AS3 v4i_ld*)(void*)&ldsV[j*64 + c8*8], 0, 0);
    }
    __builtin_amdgcn_s_wait_asynccnt(0);
    __syncthreads();
#else
#pragma unroll
    for (int it = 0; it < 8; ++it) {
      int linear = it*32 + lane;       // 0..255
      int j  = linear >> 3;            // 0..31
      int c8 = linear & 7;             // 8-element chunk of d
      const unsigned short* vp =
          kvb + ((size_t)(jb + j) * B_ + b) * KVW + D_ + n*DH_ + c8*8;
      uint4 ld = *(const uint4*)vp;
      int d0 = c8 * 8;
      ldsVt[(d0+0)*32 + j] = (unsigned short)(ld.x & 0xFFFFu);
      ldsVt[(d0+1)*32 + j] = (unsigned short)(ld.x >> 16);
      ldsVt[(d0+2)*32 + j] = (unsigned short)(ld.y & 0xFFFFu);
      ldsVt[(d0+3)*32 + j] = (unsigned short)(ld.y >> 16);
      ldsVt[(d0+4)*32 + j] = (unsigned short)(ld.z & 0xFFFFu);
      ldsVt[(d0+5)*32 + j] = (unsigned short)(ld.z >> 16);
      ldsVt[(d0+6)*32 + j] = (unsigned short)(ld.w & 0xFFFFu);
      ldsVt[(d0+7)*32 + j] = (unsigned short)(ld.w >> 16);
    }
#endif

    // --- scores: two 16x16 tiles (j and j+16)
    v8f sc[2];
#pragma unroll
    for (int jt = 0; jt < 2; ++jt) {
#pragma unroll
      for (int r = 0; r < 8; ++r) sc[jt][r] = 0.0f;
      int j = jb + jt*16 + l16;
      const unsigned short* kp = kvb + ((size_t)j * B_ + b) * KVW + n*DH_ + hi*16;
#pragma unroll
      for (int c = 0; c < 2; ++c) {
        v16bf kB = *(const v16bf*)(kp + c*32);
        sc[jt] = __builtin_amdgcn_wmma_f32_16x16x32_bf16(
            false, qA[c], false, kB, (short)0, sc[jt], false, false);
      }
#pragma unroll
      for (int r = 0; r < 8; ++r) {
        int row = i0 + hi8 + r;
        int col = jb + jt*16 + l16;
        float v = sc[jt][r] * scale;
        if (col > row) v = -1e30f;     // causal mask
        sc[jt][r] = v;
      }
    }

    // --- online softmax stats + write P (bf16) to LDS
#pragma unroll
    for (int r = 0; r < 8; ++r) {
      float mx = fmaxf(sc[0][r], sc[1][r]);
      for (int off = 8; off > 0; off >>= 1) mx = fmaxf(mx, __shfl_xor(mx, off, 32));
      float mnew = fmaxf(mrow[r], mx);
      float a = __expf(mrow[r] - mnew);
      mrow[r] = mnew;
      float p0 = __expf(sc[0][r] - mnew);
      float p1 = __expf(sc[1][r] - mnew);
      float ps = p0 + p1;
      for (int off = 8; off > 0; off >>= 1) ps += __shfl_xor(ps, off, 32);
      lrow[r] = lrow[r] * a + ps;
#pragma unroll
      for (int u = 0; u < 4; ++u) o[u][r] *= a;
      int ir = hi8 + r;
      ldsP[ir*32 + l16]      = f2bf(p0);
      ldsP[ir*32 + 16 + l16] = f2bf(p1);
    }
    __syncthreads();

    // --- P fragment (A-layout) + PV WMMAs
    v16bf pA;
    {
      const unsigned short* pp = &ldsP[l16*32 + hi*16];
      v8bf lo = *(const v8bf*)pp;
      v8bf hh = *(const v8bf*)(pp + 8);
      pA = cat8(lo, hh);
    }
#pragma unroll
    for (int u = 0; u < 4; ++u) {
      v16bf vB;
#if USE_ASYNC_LDS
#pragma unroll
      for (int e = 0; e < 16; ++e)
        vB[e] = __builtin_bit_cast(bf16_t, ldsV[(hi*16 + e)*64 + u*16 + l16]);
#else
      vB = *(const v16bf*)&ldsVt[(u*16 + l16)*32 + hi*16];
#endif
      o[u] = __builtin_amdgcn_wmma_f32_16x16x32_bf16(
          false, pA, false, vB, (short)0, o[u], false, false);
    }
  }

#pragma unroll
  for (int u = 0; u < 4; ++u)
#pragma unroll
    for (int r = 0; r < 8; ++r) {
      int row = i0 + hi8 + r;
      attn[((size_t)row * B_ + b) * D_ + n*DH_ + u*16 + l16] = o[u][r] / lrow[r];
    }
}

// ---------------------------------------------------------------- elu+1 feature maps (cQ from q, cK from v)
__global__ void elu_feat_kernel(const float* __restrict__ qf, const float* __restrict__ kvf,
                                float* __restrict__ cq, float* __restrict__ ck) {
  int i = blockIdx.x * 256 + threadIdx.x;
  if (i >= MROWS * D_) return;
  int rb = i / D_, c = i % D_;
  cq[i] = elu1(qf[i]);
  ck[i] = elu1(kvf[(size_t)rb * KVW + D_ + c]);   // NOTE reference swap: K = v
}

// ---------------------------------------------------------------- memory retrieval, delta, gated combine
__global__ void retrieve_gate_kernel(const float* __restrict__ cq, const float* __restrict__ ck,
                                     const float* __restrict__ attnf,
                                     const float* __restrict__ mem, const float* __restrict__ nrm,
                                     float* __restrict__ delta, unsigned short* __restrict__ combb) {
  int i = blockIdx.x * 256 + threadIdx.x;        // over S*B*NH*DH
  if (i >= MROWS * D_) return;
  int dv = i & 63;
  int n  = (i >> 6) % NH_;
  int rb = i / (64 * NH_);                        // s*B + b
  int b  = rb & (B_ - 1);
  const float* cqr  = cq + (size_t)rb * D_ + n*64;
  const float* ckr  = ck + (size_t)rb * D_ + n*64;
  const float* memp = mem + (((size_t)b * NH_ + n) * 64) * 64 + dv;
  const float* nrp  = nrm + ((size_t)b * NH_ + n) * 64;
  float num = 0.f, den = 0.f, numK = 0.f, denK = 0.f;
#pragma unroll 4
  for (int k = 0; k < 64; ++k) {
    float mv = memp[k * 64];
    float nv = nrp[k];
    float qv = cqr[k], kv = ckr[k];
    num  += qv * mv;  den  += qv * nv;
    numK += kv * mv;  denK += kv * nv;
  }
  float content = num / den;
  float dl = numK / denK;
  size_t oi = (size_t)rb * D_ + n*64 + dv;
  delta[oi] = dl;
  const float gate = 0.502499979f;               // sigmoid(0.01)
  combb[oi] = f2bf(gate * content + (1.0f - gate) * attnf[oi]);
}

// ---------------------------------------------------------------- memory update: mem += cK^T (k - delta)
__global__ void mem_update_kernel(const float* __restrict__ ck, const float* __restrict__ kvf,
                                  const float* __restrict__ delta, const float* __restrict__ mem,
                                  float* __restrict__ memOut) {
  int i = blockIdx.x * 256 + threadIdx.x;        // over B*NH*DH*DH = 196608
  if (i >= B_ * NH_ * DH_ * DH_) return;
  int dv = i & 63;
  int k  = (i >> 6) & 63;
  int n  = (i >> 12) % NH_;
  int b  = i / (64 * 64 * NH_);
  float acc = 0.f;
  for (int s = 0; s < S_; ++s) {
    int rb = s * B_ + b;
    float c  = ck[(size_t)rb * D_ + n*64 + k];
    float vv = kvf[(size_t)rb * KVW + n*64 + dv];   // "Vb" = k tensor (first half of kv)
    float dl = delta[(size_t)rb * D_ + n*64 + dv];
    acc += c * (vv - dl);
  }
  memOut[i] = mem[i] + acc;
}

// ---------------------------------------------------------------- norm update: norm += sum_s cK
__global__ void norm_update_kernel(const float* __restrict__ ck, const float* __restrict__ nrm,
                                   float* __restrict__ nrmOut) {
  int i = blockIdx.x * 256 + threadIdx.x;        // over B*NH*DH = 3072
  if (i >= B_ * NH_ * DH_) return;
  int k = i & 63;
  int n = (i >> 6) % NH_;
  int b = i / (64 * NH_);
  float acc = 0.f;
  for (int s = 0; s < S_; ++s)
    acc += ck[(size_t)(s * B_ + b) * D_ + n*64 + k];
  nrmOut[i] = nrm[i] + acc;
}

// ---------------------------------------------------------------- residual + LayerNorm (per row of 768)
__global__ void ln_residual_kernel(const float* __restrict__ x, const float* __restrict__ y,
                                   const float* __restrict__ g, const float* __restrict__ bb,
                                   float* __restrict__ outf, unsigned short* __restrict__ outb) {
  __shared__ float red[256];
  const int row = blockIdx.x, t = threadIdx.x;
  const float* xr = x + (size_t)row * D_;
  const float* yr = y + (size_t)row * D_;
  float v[3]; float s = 0.f;
#pragma unroll
  for (int e = 0; e < 3; ++e) { v[e] = xr[t + e*256] + yr[t + e*256]; s += v[e]; }
  red[t] = s; __syncthreads();
  for (int off = 128; off > 0; off >>= 1) { if (t < off) red[t] += red[t + off]; __syncthreads(); }
  float mu = red[0] / (float)D_;
  __syncthreads();
  s = 0.f;
#pragma unroll
  for (int e = 0; e < 3; ++e) { float d = v[e] - mu; s += d * d; }
  red[t] = s; __syncthreads();
  for (int off = 128; off > 0; off >>= 1) { if (t < off) red[t] += red[t + off]; __syncthreads(); }
  float inv = rsqrtf(red[0] / (float)D_ + 1e-5f);
#pragma unroll
  for (int e = 0; e < 3; ++e) {
    int c = t + e*256;
    float o = (v[e] - mu) * inv * g[c] + bb[c];
    outf[(size_t)row * D_ + c] = o;
    if (outb) outb[(size_t)row * D_ + c] = f2bf(o);
  }
}

// ================================================================ host
extern "C" void kernel_launch(void* const* d_in, const int* in_sizes, int n_in,
                              void* d_out, int out_size, void* d_ws, size_t ws_size,
                              hipStream_t stream) {
  (void)in_sizes; (void)n_in; (void)out_size; (void)ws_size;
  const int*   inp  = (const int*)  d_in[0];
  const float* we   = (const float*)d_in[1];
  const float* Wq   = (const float*)d_in[2];
  const float* Wkv  = (const float*)d_in[3];
  const float* Wo   = (const float*)d_in[4];
  const float* ln1g = (const float*)d_in[5];
  const float* ln1b = (const float*)d_in[6];
  const float* fW1  = (const float*)d_in[7];
  const float* fb1  = (const float*)d_in[8];
  const float* fW2  = (const float*)d_in[9];
  const float* fb2  = (const float*)d_in[10];
  const float* ln2g = (const float*)d_in[11];
  const float* ln2b = (const float*)d_in[12];
  const float* mem  = (const float*)d_in[13];
  const float* nrm  = (const float*)d_in[14];

  float* out      = (float*)d_out;
  float* h_out    = out;
  float* mems_out = out + (size_t)MROWS * D_;
  float* nrms_out = mems_out + (size_t)L_ * B_ * NH_ * DH_ * DH_;

  size_t off = 0;
  auto take = [&](size_t bytes) -> void* {
    void* p = (void*)((char*)d_ws + off);
    off += (bytes + 255) & ~(size_t)255;
    return p;
  };
  const size_t SBD = (size_t)MROWS * D_;
  float*          hf    = (float*)         take(SBD * 4);
  unsigned short* hb    = (unsigned short*)take(SBD * 2);
  float*          qf    = (float*)         take(SBD * 4);
  unsigned short* qb    = (unsigned short*)take(SBD * 2);
  float*          kvf   = (float*)         take((size_t)MROWS * KVW * 4);
  unsigned short* kvb   = (unsigned short*)take((size_t)MROWS * KVW * 2);
  float*          attnf = (float*)         take(SBD * 4);
  unsigned short* combb = (unsigned short*)take(SBD * 2);
  float*          projf = (float*)         take(SBD * 4);
  float*          out1f = (float*)         take(SBD * 4);
  unsigned short* out1b = (unsigned short*)take(SBD * 2);
  unsigned short* ff1b  = (unsigned short*)take((size_t)MROWS * DI_ * 2);
  float*          cq    = (float*)         take(SBD * 4);
  float*          ck    = (float*)         take(SBD * 4);
  float*          delta = (float*)         take(SBD * 4);
  unsigned short* wqB   = (unsigned short*)take((size_t)D_ * D_ * 2);
  unsigned short* wkvB  = (unsigned short*)take((size_t)KVW * D_ * 2);
  unsigned short* woB   = (unsigned short*)take((size_t)D_ * D_ * 2);
  unsigned short* w1B   = (unsigned short*)take((size_t)DI_ * D_ * 2);
  unsigned short* w2B   = (unsigned short*)take((size_t)D_ * DI_ * 2);

  const int elemGrid = (int)((SBD + 255) / 256);
  embed_kernel<<<elemGrid, 256, 0, stream>>>(inp, we, hf, hb);

  for (int l = 0; l < L_; ++l) {
    // weight conversion (f32 -> bf16)
    auto cast = [&](const float* s, unsigned short* d, int n) {
      cast_bf16_kernel<<<(n + 255) / 256, 256, 0, stream>>>(s, d, n);
    };
    cast(Wq  + (size_t)l * D_ * D_,  wqB,  D_ * D_);
    cast(Wkv + (size_t)l * KVW * D_, wkvB, KVW * D_);
    cast(Wo  + (size_t)l * D_ * D_,  woB,  D_ * D_);
    cast(fW1 + (size_t)l * DI_ * D_, w1B,  DI_ * D_);
    cast(fW2 + (size_t)l * D_ * DI_, w2B,  D_ * DI_);

    // q = h @ Wq^T ; kv = h @ Wkv^T
    gemm_bf16_wmma<<<dim3(D_/64,  MROWS/32), 32, 0, stream>>>(hb, wqB,  nullptr, qf,  qb,  MROWS, D_,  D_, 0);
    gemm_bf16_wmma<<<dim3(KVW/64, MROWS/32), 32, 0, stream>>>(hb, wkvB, nullptr, kvf, kvb, MROWS, KVW, D_, 0);

    // causal flash attention
    flash_attn_wmma<<<dim3(S_/16, B_, NH_), 32, 0, stream>>>(qb, kvb, attnf);

    // InfiniAttention memory path
    const float* memL = mem + (size_t)l * B_ * NH_ * DH_ * DH_;
    const float* nrmL = nrm + (size_t)l * B_ * NH_ * DH_;
    elu_feat_kernel<<<elemGrid, 256, 0, stream>>>(qf, kvf, cq, ck);
    retrieve_gate_kernel<<<elemGrid, 256, 0, stream>>>(cq, ck, attnf, memL, nrmL, delta, combb);
    mem_update_kernel<<<(B_*NH_*DH_*DH_ + 255)/256, 256, 0, stream>>>(
        ck, kvf, delta, memL, mems_out + (size_t)l * B_ * NH_ * DH_ * DH_);
    norm_update_kernel<<<(B_*NH_*DH_ + 255)/256, 256, 0, stream>>>(
        ck, nrmL, nrms_out + (size_t)l * B_ * NH_ * DH_);

    // output projection + LN1
    gemm_bf16_wmma<<<dim3(D_/64, MROWS/32), 32, 0, stream>>>(combb, woB, nullptr, projf, nullptr, MROWS, D_, D_, 0);
    ln_residual_kernel<<<MROWS, 256, 0, stream>>>(hf, projf, ln1g + (size_t)l*D_, ln1b + (size_t)l*D_, out1f, out1b);

    // FFN
    gemm_bf16_wmma<<<dim3(DI_/64, MROWS/32), 32, 0, stream>>>(out1b, w1B, fb1 + (size_t)l*DI_, nullptr, ff1b, MROWS, DI_, D_, 1);
    gemm_bf16_wmma<<<dim3(D_/64,  MROWS/32), 32, 0, stream>>>(ff1b,  w2B, fb2 + (size_t)l*D_,  projf, nullptr, MROWS, D_, DI_, 0);

    // LN2 -> next h (last layer writes straight to d_out)
    float* hNext = (l == L_ - 1) ? h_out : hf;
    ln_residual_kernel<<<MROWS, 256, 0, stream>>>(out1f, projf, ln2g + (size_t)l*D_, ln2b + (size_t)l*D_, hNext, hb);
  }
}
